// MultiHeadAttention_29712583753823
// MI455X (gfx1250) — matrix operand order repaired and checked by reading, not compiled, not run on hardware
//
#include <hip/hip_runtime.h>
#include <hip/hip_bf16.h>

// ---------------- WMMA fragment types (gfx1250, wave32) ----------------
typedef __attribute__((ext_vector_type(16))) __bf16 v16bf;
typedef __attribute__((ext_vector_type(8)))  __bf16 v8bf;
typedef __attribute__((ext_vector_type(8)))  float  v8f;

union FragU { v16bf v; unsigned short u[16]; };

__device__ __forceinline__ unsigned short f32_to_bf16u(float f) {
    unsigned u = __float_as_uint(f);
    unsigned r = u + 0x7FFFu + ((u >> 16) & 1u);      // round-to-nearest-even
    if ((u & 0x7F800000u) == 0x7F800000u) r = u;      // Inf/NaN passthrough
    return (unsigned short)(r >> 16);
}
__device__ __forceinline__ float bf16u_to_f32(unsigned short h) {
    return __uint_as_float(((unsigned)h) << 16);
}

// A-matrix 16x32 bf16: lane m = lane&15; contiguous K chunk of 16 by lane>>4:
//   elems 0..15 -> K = kg*16 .. kg*16+15
__device__ __forceinline__ v16bf load_frag_a(const __bf16* p, int stride, int lane) {
    const __bf16* r = p + (size_t)(lane & 15) * stride + ((lane >> 4) & 1) * 16;
    v8bf lo = *(const v8bf*)(r);
    v8bf hi = *(const v8bf*)(r + 8);
    return __builtin_shufflevector(lo, hi, 0,1,2,3,4,5,6,7,8,9,10,11,12,13,14,15);
}
// B-matrix 32x16 bf16: lane holds column n = lane&15, contiguous K chunk of 16
// selected by lane>>4.  Source stored N-major with contiguous K (i.e. W rows).
__device__ __forceinline__ v16bf load_frag_b(const __bf16* p, int stride, int lane) {
    const __bf16* r = p + (size_t)(lane & 15) * stride + ((lane >> 4) & 1) * 16;
    return *(const v16bf*)(r);
}
__device__ __forceinline__ v8f wmma_bf16(v16bf a, v16bf b, v8f c) {
    return __builtin_amdgcn_wmma_f32_16x16x32_bf16(false, a, false, b, (short)0, c,
                                                   false, false);
}

// ---------------- Kernel 1: f32 -> bf16 conversion ----------------
__global__ void mha_cvt_bf16_kernel(const float* __restrict__ src,
                                    unsigned short* __restrict__ dst, int n) {
    int i = blockIdx.x * blockDim.x + threadIdx.x;
    if (i < n) dst[i] = f32_to_bf16u(src[i]);
}

// ---------------- Kernel 2/4: GEMM  Y = X (M=4096 x K=1024) * W^T + bias ----
// W stored (N=1024, K=1024) row-major -> directly B-matrix friendly.
// mode 0: head-split bf16  out[b,h,t,d]          (scale applied; Q uses 1/8)
// mode 2: head-split transposed bf16 out[b,h,d,t] (for V)
// mode 3: plain f32 row-major (final projection)
__global__ __launch_bounds__(256) void mha_gemm_bf16_kernel(
    const __bf16* __restrict__ X, const __bf16* __restrict__ W,
    const float* __restrict__ bias,
    unsigned short* __restrict__ out_bf, float* __restrict__ out_f32,
    float scale, int mode)
{
    const int K = 1024;
    int lane = threadIdx.x & 31;
    int wave = threadIdx.x >> 5;
    int m0 = blockIdx.y * 128 + wave * 16;
    int n0 = blockIdx.x * 64;

    const __bf16* arow = X + (size_t)m0 * K;
    const __bf16* w0 = W + (size_t)(n0 +  0) * K;
    const __bf16* w1 = W + (size_t)(n0 + 16) * K;
    const __bf16* w2 = W + (size_t)(n0 + 32) * K;
    const __bf16* w3 = W + (size_t)(n0 + 48) * K;

    v8f acc0 = {}, acc1 = {}, acc2 = {}, acc3 = {};
    for (int k0 = 0; k0 < K; k0 += 32) {
        __builtin_prefetch((const void*)(arow + (size_t)(lane & 15) * K + k0 + 256), 0, 1);
        v16bf a  = load_frag_a(arow + k0, K, lane);
        v16bf b0 = load_frag_b(w0 + k0, K, lane);
        v16bf b1 = load_frag_b(w1 + k0, K, lane);
        v16bf b2 = load_frag_b(w2 + k0, K, lane);
        v16bf b3 = load_frag_b(w3 + k0, K, lane);
        acc0 = wmma_bf16(a, b0, acc0);
        acc1 = wmma_bf16(a, b1, acc1);
        acc2 = wmma_bf16(a, b2, acc2);
        acc3 = wmma_bf16(a, b3, acc3);
    }
    v8f accs[4] = {acc0, acc1, acc2, acc3};
    int kg = lane >> 4, nl = lane & 15;
    for (int j = 0; j < 4; ++j) {
        for (int r = 0; r < 8; ++r) {
            int m = m0 + r + 8 * kg;              // C/D layout: row = vgpr + 8*(lane>=16)
            int n = n0 + j * 16 + nl;             //            col = lane&15
            float v = (accs[j][r] + bias[n]) * scale;
            int b = m >> 10, t = m & 1023, h = n >> 6, d = n & 63;
            if (mode == 0)
                out_bf[(((size_t)(b * 16 + h) << 10) + t) * 64 + d] = f32_to_bf16u(v);
            else if (mode == 2)
                out_bf[(((size_t)(b * 16 + h) * 64 + d) << 10) + t] = f32_to_bf16u(v);
            else
                out_f32[(size_t)m * 1024 + n] = v;
        }
    }
}

// ---------------- Kernel 3: fused attention ----------------
// Per block: one (b,h) and QB=32 query rows.  8 waves / 256 threads.
#define QB 32
__global__ __launch_bounds__(256) void mha_attn_kernel(
    const __bf16* __restrict__ qh,   // (B*H, T, 64) bf16, pre-scaled 1/8
    const __bf16* __restrict__ kh,   // (B*H, T, 64) bf16
    const __bf16* __restrict__ vT,   // (B*H, 64, T) bf16
    const float*  __restrict__ rpe,  // (33, 128) f32
    float* __restrict__ align_out,   // (B*H, T, T) f32
    unsigned short* __restrict__ ctx_out) // (B, T, DIM) bf16
{
    __shared__ float sc[QB][1024];   // scores -> probabilities (128 KB)
    __shared__ float pv[QB][33];     // qh . rpe_q per distance bucket
    __shared__ float sbuck[QB][33];  // prob mass per distance bucket
    __shared__ float crpe[QB][64];   // rpe_v context contribution

    int tid  = threadIdx.x;
    int lane = tid & 31;
    int wave = tid >> 5;
    int bh = blockIdx.y;             // b*16 + h
    int b = bh >> 4, h = bh & 15;
    int q0 = blockIdx.x * QB;
    const unsigned short* qh_u = (const unsigned short*)qh;

    // Phase 0: pv[q][d33] = dot(qh[q], rpe[d33, 0:64])
    for (int idx = tid; idx < QB * 33; idx += 256) {
        int qr = idx / 33, d33 = idx % 33;
        const unsigned short* qp = qh_u + (((size_t)bh << 10) + q0 + qr) * 64;
        float s = 0.f;
        for (int j = 0; j < 64; ++j) s += bf16u_to_f32(qp[j]) * rpe[d33 * 128 + j];
        pv[qr][d33] = s;
    }
    __syncthreads();

    // Phase 1: energy = qh @ kh^T (+pv) via WMMA, into LDS
    {
        int qt  = wave >> 2;                 // 0..1 : 16-row q subtile
        int kt0 = (wave & 3) * 16;           // 16 k-tiles per wave
        const __bf16* qbase = qh + (((size_t)bh << 10) + q0 + qt * 16) * 64;
        v16bf a0 = load_frag_a(qbase + 0,  64, lane);
        v16bf a1 = load_frag_a(qbase + 32, 64, lane);
        int kg = lane >> 4, nl = lane & 15;
        for (int kt = kt0; kt < kt0 + 16; ++kt) {
            const __bf16* kbase = kh + (((size_t)bh << 10) + kt * 16) * 64;
            v16bf b0 = load_frag_b(kbase + 0,  64, lane);
            v16bf b1 = load_frag_b(kbase + 32, 64, lane);
            v8f e = {};
            e = wmma_bf16(a0, b0, e);
            e = wmma_bf16(a1, b1, e);
            int kcol = kt * 16 + nl;
            for (int r = 0; r < 8; ++r) {
                int qr = qt * 16 + r + 8 * kg;
                int dist = (q0 + qr) - kcol;
                dist = dist < -16 ? -16 : (dist > 16 ? 16 : dist);
                sc[qr][kcol] = e[r] + pv[qr][dist + 16];
            }
        }
    }
    __syncthreads();

    // Phase 2a: softmax per row (8 threads/row, chunks of 128), + bucket sums
    {
        int row = tid >> 3, sub = tid & 7;
        int ks = sub * 128;
        float m = -3.4e38f;
        for (int k = ks; k < ks + 128; ++k) m = fmaxf(m, sc[row][k]);
        m = fmaxf(m, __shfl_xor(m, 1, 32));
        m = fmaxf(m, __shfl_xor(m, 2, 32));
        m = fmaxf(m, __shfl_xor(m, 4, 32));
        float s = 0.f;
        for (int k = ks; k < ks + 128; ++k) {
            float p = __expf(sc[row][k] - m);
            sc[row][k] = p;
            s += p;
        }
        s += __shfl_xor(s, 1, 32);
        s += __shfl_xor(s, 2, 32);
        s += __shfl_xor(s, 4, 32);
        float inv = 1.0f / s;
        int qg = q0 + row;
        int lim = qg - 16;                    // inclusive: bucket 32 (dist clamp +16)
        float head = 0.f;
        for (int k = ks; k < ks + 128; ++k) {
            float p = sc[row][k] * inv;
            sc[row][k] = p;
            if (k <= lim) head += p;
        }
        head += __shfl_xor(head, 1, 32);
        head += __shfl_xor(head, 2, 32);
        head += __shfl_xor(head, 4, 32);
        if (sub == 0) {
            sbuck[row][32] = head;
            float mid = 0.f;
            for (int o = -15; o <= 15; ++o) {     // buckets 1..31 : single k
                int k = qg - o;
                float p = (k >= 0 && k < 1024) ? sc[row][k] : 0.f;
                sbuck[row][o + 16] = p;
                mid += p;
            }
            sbuck[row][0] = fmaxf(0.f, 1.0f - head - mid);  // tail: rows sum to 1
        }
    }
    __syncthreads();

    // Phase 2b: coalesced alignment write (f32, one pass to HBM)
    {
        float* abase = align_out + (((size_t)bh << 10) + q0) * 1024;
        for (int idx = tid; idx < QB * 1024; idx += 256)
            abase[idx] = sc[idx >> 10][idx & 1023];
    }
    // Phase 2c: crpe[q][d] = sum_d33 sbuck[q][d33] * rpe[d33, 64+d]
    for (int idx = tid; idx < QB * 64; idx += 256) {
        int qr = idx >> 6, d = idx & 63;
        float s = 0.f;
        for (int d33 = 0; d33 < 33; ++d33)
            s += sbuck[qr][d33] * rpe[d33 * 128 + 64 + d];
        crpe[qr][d] = s;
    }
    __syncthreads();

    // Phase 3: context = probs @ vh (+ crpe) via WMMA; A from LDS probs
    {
        int qt = wave >> 2;                   // 0..1
        int dt = wave & 3;                    // 0..3 (16 head-dims each)
        int kg = lane >> 4, nl = lane & 15;
        v8f acc = {};
        const __bf16* vbase = vT + ((size_t)bh * 64 + dt * 16 + nl) * 1024 + kg * 16;
        for (int k0 = 0; k0 < 1024; k0 += 32) {
            const float* prow = &sc[qt * 16 + nl][k0 + kg * 16];
            FragU fa;
            for (int i = 0; i < 8; ++i) fa.u[i]     = f32_to_bf16u(prow[i]);
            for (int i = 0; i < 8; ++i) fa.u[8 + i] = f32_to_bf16u(prow[8 + i]);
            v16bf bfr = *(const v16bf*)(vbase + k0);
            acc = wmma_bf16(fa.v, bfr, acc);
        }
        for (int r = 0; r < 8; ++r) {
            int qr = qt * 16 + r + 8 * kg;
            int d  = dt * 16 + nl;
            float v = acc[r] + crpe[qr][d];
            size_t off = (((size_t)b << 10) + (q0 + qr)) * 1024 + h * 64 + d;
            ctx_out[off] = f32_to_bf16u(v);
        }
    }
}

// ---------------- Host launch ----------------
extern "C" void kernel_launch(void* const* d_in, const int* in_sizes, int n_in,
                              void* d_out, int out_size, void* d_ws, size_t ws_size,
                              hipStream_t stream) {
    const float* q   = (const float*)d_in[0];
    const float* k   = (const float*)d_in[1];
    const float* v   = (const float*)d_in[2];
    const float* Wq  = (const float*)d_in[3];
    const float* bq  = (const float*)d_in[4];
    const float* Wk  = (const float*)d_in[5];
    const float* bk  = (const float*)d_in[6];
    const float* Wv  = (const float*)d_in[7];
    const float* bv  = (const float*)d_in[8];
    const float* rpe = (const float*)d_in[9];
    const float* Wo  = (const float*)d_in[10];
    const float* bo  = (const float*)d_in[11];

    char* ws = (char*)d_ws;
    const size_t MB = 1ull << 20;
    unsigned short* qbf  = (unsigned short*)(ws +  0 * MB);  // 8 MB each (4M elems)
    unsigned short* kbf  = (unsigned short*)(ws +  8 * MB);
    unsigned short* vbf  = (unsigned short*)(ws + 16 * MB);
    unsigned short* Wqbf = (unsigned short*)(ws + 24 * MB);  // 2 MB each
    unsigned short* Wkbf = (unsigned short*)(ws + 26 * MB);
    unsigned short* Wvbf = (unsigned short*)(ws + 28 * MB);
    unsigned short* Wobf = (unsigned short*)(ws + 30 * MB);
    unsigned short* qhb  = (unsigned short*)(ws + 32 * MB);  // 8 MB each
    unsigned short* khb  = (unsigned short*)(ws + 40 * MB);
    unsigned short* vTb  = (unsigned short*)(ws + 48 * MB);
    unsigned short* ctxb = (unsigned short*)(ws + 56 * MB);

    const int nBig = 4 * 1024 * 1024;
    const int nW   = 1024 * 1024;
    mha_cvt_bf16_kernel<<<(nBig + 255) / 256, 256, 0, stream>>>(q,  qbf,  nBig);
    mha_cvt_bf16_kernel<<<(nBig + 255) / 256, 256, 0, stream>>>(k,  kbf,  nBig);
    mha_cvt_bf16_kernel<<<(nBig + 255) / 256, 256, 0, stream>>>(v,  vbf,  nBig);
    mha_cvt_bf16_kernel<<<(nW   + 255) / 256, 256, 0, stream>>>(Wq, Wqbf, nW);
    mha_cvt_bf16_kernel<<<(nW   + 255) / 256, 256, 0, stream>>>(Wk, Wkbf, nW);
    mha_cvt_bf16_kernel<<<(nW   + 255) / 256, 256, 0, stream>>>(Wv, Wvbf, nW);
    mha_cvt_bf16_kernel<<<(nW   + 255) / 256, 256, 0, stream>>>(Wo, Wobf, nW);

    dim3 gg(16, 32), gb(256);   // N/64 x M/128
    mha_gemm_bf16_kernel<<<gg, gb, 0, stream>>>((const __bf16*)qbf, (const __bf16*)Wqbf,
                                                bq, qhb, nullptr, 0.125f, 0);
    mha_gemm_bf16_kernel<<<gg, gb, 0, stream>>>((const __bf16*)kbf, (const __bf16*)Wkbf,
                                                bk, khb, nullptr, 1.0f, 0);
    mha_gemm_bf16_kernel<<<gg, gb, 0, stream>>>((const __bf16*)vbf, (const __bf16*)Wvbf,
                                                bv, vTb, nullptr, 1.0f, 2);

    float* out_f  = (float*)d_out;
    float* align  = out_f + (size_t)4 * 1024 * 1024;
    mha_attn_kernel<<<dim3(32, 64), 256, 0, stream>>>(
        (const __bf16*)qhb, (const __bf16*)khb, (const __bf16*)vTb, rpe, align, ctxb);

    mha_gemm_bf16_kernel<<<gg, gb, 0, stream>>>((const __bf16*)ctxb, (const __bf16*)Wobf,
                                                bo, nullptr, out_f, 1.0f, 3);
}
